// PatchedDeepseekV2Attention_14645838479469
// MI455X (gfx1250) — matrix operand order repaired and checked by reading, not compiled, not run on hardware
//
#include <hip/hip_runtime.h>
#include <stdint.h>

#define T_SEQ 2048
#define HID   5120
#define QR    1536
#define KVR   512
#define DR    64
#define DN    128
#define DV    128
#define NH    32
#define QH    192          // DN + DR
#define DF    576          // KVR + DR
#define ODIM  (NH * DV)    // 4096
#define QBD   (NH * QH)    // 6144

typedef __bf16 v16bf __attribute__((ext_vector_type(16)));
typedef float  v8f   __attribute__((ext_vector_type(8)));
typedef unsigned int u32x4 __attribute__((ext_vector_type(4)));

union ABFrag { v16bf v; u32x4 q[2]; };

__device__ __forceinline__ unsigned short f32_to_bf16(float f) {
  union { float f; unsigned int u; } c; c.f = f;
  unsigned int u = c.u;
  return (unsigned short)((u + 0x7FFFu + ((u >> 16) & 1u)) >> 16);
}

__device__ __forceinline__ v8f wmma_bf16(v16bf a, v16bf b, v8f c) {
  return __builtin_amdgcn_wmma_f32_16x16x32_bf16(false, a, false, b, (short)0, c, false, false);
}

// ---------------------------------------------------------------------------
// Generic NT GEMM: C[M,N] = A[M,K] * B[N,K]^T, bf16 inputs, f32 accumulate.
// OUT_MODE 0: f32 C[row*ldc+col]; 1: bf16 C[row*ldc+col]; 2: bf16 C[col*ldc+row]
// M is a multiple of 128 (covered by grid.y); K a multiple of 32; N clamped.
// ---------------------------------------------------------------------------
template <int OUT_MODE>
__global__ __launch_bounds__(256) void gemm_nt_kernel(
    const unsigned short* __restrict__ A, int lda, long long strideA,
    const unsigned short* __restrict__ B, int ldb, long long strideB,
    void* __restrict__ Cvoid, int ldc, long long strideC,
    int N, int K)
{
  const int bz = blockIdx.z;
  const unsigned short* Ab = A + (long long)bz * strideA;
  const unsigned short* Bb = B + (long long)bz * strideB;
  const int lane  = threadIdx.x & 31;
  const int wave  = threadIdx.x >> 5;
  const int lmod  = lane & 15;
  const int lhalf = lane >> 4;
  const int m0 = blockIdx.y * 128 + (wave & 1) * 64;
  const int n0 = blockIdx.x * 128 + (wave >> 1) * 32;

  v8f acc[4][2];
  for (int i = 0; i < 4; ++i)
    for (int j = 0; j < 2; ++j) acc[i][j] = 0.0f;

  // A fragment: lanes 0-15 rows m, K = kb..kb+7 (q0) and kb+16..kb+23 (q1);
  //             lanes 16-31 K = kb+8.. and kb+24..  -> base offset lhalf*8.
  const unsigned short* arow[4];
  for (int mt = 0; mt < 4; ++mt)
    arow[mt] = Ab + (long long)(m0 + mt * 16 + lmod) * lda + lhalf * 8;
  // B fragment (from W[n,k]): same K split as A (lhalf*8 base, q1 at +16).
  const unsigned short* brow[2];
  for (int nt = 0; nt < 2; ++nt) {
    int n = n0 + nt * 16 + lmod;
    if (n > N - 1) n = N - 1;            // clamp (stores are guarded)
    brow[nt] = Bb + (long long)n * ldb + lhalf * 8;
  }

  for (int kb = 0; kb < K; kb += 32) {
    ABFrag bfr[2];
    for (int nt = 0; nt < 2; ++nt) {
      bfr[nt].q[0] = *(const u32x4*)(brow[nt] + kb);
      bfr[nt].q[1] = *(const u32x4*)(brow[nt] + kb + 16);
    }
    for (int mt = 0; mt < 4; ++mt) {
      ABFrag af;
      af.q[0] = *(const u32x4*)(arow[mt] + kb);
      af.q[1] = *(const u32x4*)(arow[mt] + kb + 16);
      acc[mt][0] = wmma_bf16(af.v, bfr[0].v, acc[mt][0]);
      acc[mt][1] = wmma_bf16(af.v, bfr[1].v, acc[mt][1]);
    }
  }

  for (int mt = 0; mt < 4; ++mt)
    for (int nt = 0; nt < 2; ++nt) {
      int col = n0 + nt * 16 + lmod;
      if (col >= N) continue;
      for (int r = 0; r < 8; ++r) {
        int row = m0 + mt * 16 + lhalf * 8 + r;     // C layout: half-wave row split
        float v = acc[mt][nt][r];
        if (OUT_MODE == 0)
          ((float*)Cvoid + (long long)bz * strideC)[(long long)row * ldc + col] = v;
        else if (OUT_MODE == 1)
          ((unsigned short*)Cvoid + (long long)bz * strideC)[(long long)row * ldc + col] = f32_to_bf16(v);
        else
          ((unsigned short*)Cvoid + (long long)bz * strideC)[(long long)col * ldc + row] = f32_to_bf16(v);
      }
    }
}

// ---------------------------------------------------------------------------
// Flash attention, causal.  grid = (T/32, H), block = 64 (2 waves, 16 rows/wave)
// qf: [H][T][DF] bf16, kf: [T][DF] bf16, vt: [H][DV][T] bf16, obf: [T][ODIM] bf16
// ---------------------------------------------------------------------------
#define LQ_STRIDE 584   // 576 + 8 pad -> conflict-free ds_load_b128
#define LP_STRIDE 40    // 32 + 8 pad

__global__ __launch_bounds__(64) void attn_kernel(
    const unsigned short* __restrict__ qf,
    const unsigned short* __restrict__ kf,
    const unsigned short* __restrict__ vt,
    unsigned short* __restrict__ obf)
{
  __shared__ unsigned short lds_q[32 * LQ_STRIDE];
  __shared__ unsigned short lds_p[2][16 * LP_STRIDE];

  const int head  = blockIdx.y;
  const int qbase = blockIdx.x * 32;
  const int tid   = threadIdx.x;
  const int lane  = tid & 31;
  const int wave  = tid >> 5;
  const int lmod  = lane & 15;
  const int lhalf = lane >> 4;
  const float SC  = 0.07216878364870323f;   // 192^-0.5

  const unsigned short* qhead = qf + (long long)head * T_SEQ * DF;
  // Stage 32 q-rows into LDS via async L2->LDS DMA (b128 per lane, ASYNCcnt).
  for (int idx = tid; idx < 32 * (DF / 8); idx += 64) {
    int row = idx / (DF / 8), ch = idx % (DF / 8);
    unsigned lds_off =
        (unsigned)(uintptr_t)&lds_q[row * LQ_STRIDE + ch * 8];
    unsigned long long ga = (unsigned long long)(uintptr_t)
        (qhead + (long long)(qbase + row) * DF + ch * 8);
    asm volatile("global_load_async_to_lds_b128 %0, %1, off"
                 :: "v"(lds_off), "v"(ga) : "memory");
  }
  asm volatile("s_wait_asynccnt 0x0" ::: "memory");
  __syncthreads();

  const int row_min = qbase + wave * 16;
  const unsigned short* ldsA = &lds_q[(wave * 16 + lmod) * LQ_STRIDE + lhalf * 8];
  const unsigned short* vhead = vt + (long long)head * DV * T_SEQ;

  v8f acc[8];
  for (int i = 0; i < 8; ++i) acc[i] = 0.0f;
  float mrow[8], lrow[8];
  for (int r = 0; r < 8; ++r) { mrow[r] = -1e30f; lrow[r] = 0.0f; }

  for (int j0 = 0; j0 <= row_min + 15; j0 += 32) {
    // Software prefetch of next key block (speculative, clamped in-bounds).
    {
      int pj0 = j0 + 32 + lmod;
      int pj1 = pj0 + 16;
      if (pj0 > T_SEQ - 1) pj0 = T_SEQ - 1;
      if (pj1 > T_SEQ - 1) pj1 = T_SEQ - 1;
      __builtin_prefetch(kf + (long long)pj0 * DF, 0, 1);
      __builtin_prefetch(kf + (long long)pj1 * DF, 0, 1);
    }
    // ---- S = Q * K^T  (2 key tiles of 16) ----
    v8f s0 = 0.0f, s1 = 0.0f;
    const unsigned short* k0 = kf + (long long)(j0 + lmod) * DF + lhalf * 8;
    const unsigned short* k1 = kf + (long long)(j0 + 16 + lmod) * DF + lhalf * 8;
    for (int kb = 0; kb < DF; kb += 32) {
      ABFrag a, b0, b1;
      a.q[0]  = *(const u32x4*)(ldsA + kb);
      a.q[1]  = *(const u32x4*)(ldsA + kb + 16);
      b0.q[0] = *(const u32x4*)(k0 + kb);
      b0.q[1] = *(const u32x4*)(k0 + kb + 16);
      b1.q[0] = *(const u32x4*)(k1 + kb);
      b1.q[1] = *(const u32x4*)(k1 + kb + 16);
      s0 = wmma_bf16(a.v, b0.v, s0);
      s1 = wmma_bf16(a.v, b1.v, s1);
    }
    // ---- scale, causal mask, online softmax ----
    const int c0 = j0 + lmod, c1 = c0 + 16;
    float alpha_r[8];
    for (int r = 0; r < 8; ++r) {
      int rowr = row_min + lhalf * 8 + r;
      float v0 = (c0 <= rowr) ? s0[r] * SC : -1e30f;
      float v1 = (c1 <= rowr) ? s1[r] * SC : -1e30f;
      float mx = fmaxf(v0, v1);
      mx = fmaxf(mx, __shfl_xor(mx, 1, 32));
      mx = fmaxf(mx, __shfl_xor(mx, 2, 32));
      mx = fmaxf(mx, __shfl_xor(mx, 4, 32));
      mx = fmaxf(mx, __shfl_xor(mx, 8, 32));
      float mnew  = fmaxf(mrow[r], mx);
      float alpha = __expf(mrow[r] - mnew);
      mrow[r] = mnew;
      float p0 = __expf(v0 - mnew);
      float p1 = __expf(v1 - mnew);
      lrow[r] = lrow[r] * alpha + p0 + p1;
      alpha_r[r] = alpha;
      int prow = lhalf * 8 + r;
      lds_p[wave][prow * LP_STRIDE + lmod]      = f32_to_bf16(p0);
      lds_p[wave][prow * LP_STRIDE + 16 + lmod] = f32_to_bf16(p1);
    }
    for (int nt = 0; nt < 8; ++nt)
      for (int r = 0; r < 8; ++r) acc[nt][r] *= alpha_r[r];
    // wave-local LDS turnaround: DS ops are in-order, force visibility
    asm volatile("s_wait_dscnt 0x0" ::: "memory");
    // ---- P (A-layout from LDS) @ V ----
    ABFrag pf;
    const unsigned short* pp = &lds_p[wave][lmod * LP_STRIDE + lhalf * 8];
    pf.q[0] = *(const u32x4*)(pp);
    pf.q[1] = *(const u32x4*)(pp + 16);
    for (int nt = 0; nt < 8; ++nt) {
      ABFrag bv;
      const unsigned short* pv = vhead + (long long)(nt * 16 + lmod) * T_SEQ + j0 + lhalf * 8;
      bv.q[0] = *(const u32x4*)(pv);
      bv.q[1] = *(const u32x4*)(pv + 16);
      acc[nt] = wmma_bf16(pf.v, bv.v, acc[nt]);
    }
  }

  // ---- epilogue: normalize, store bf16 ----
  for (int r = 0; r < 8; ++r) {
    float l = lrow[r];
    l += __shfl_xor(l, 1, 32);
    l += __shfl_xor(l, 2, 32);
    l += __shfl_xor(l, 4, 32);
    l += __shfl_xor(l, 8, 32);
    lrow[r] = 1.0f / l;
  }
  for (int nt = 0; nt < 8; ++nt)
    for (int r = 0; r < 8; ++r) {
      int row = row_min + lhalf * 8 + r;
      obf[(long long)row * ODIM + head * DV + nt * 16 + lmod] =
          f32_to_bf16(acc[nt][r] * lrow[r]);
    }
}

// ---------------------------------------------------------------------------
// Elementwise / normalization / rope / transpose helpers
// ---------------------------------------------------------------------------
__global__ void cvt_bf16_kernel(const float* __restrict__ in,
                                unsigned short* __restrict__ out, long long n) {
  long long i = (long long)blockIdx.x * blockDim.x + threadIdx.x;
  long long stride = (long long)gridDim.x * blockDim.x;
  for (; i < n; i += stride) out[i] = f32_to_bf16(in[i]);
}

// in[b][R][C] (f32) -> out[b][C][R] (bf16); R,C multiples of 32
__global__ __launch_bounds__(256) void transpose_cvt_kernel(
    const float* __restrict__ in, unsigned short* __restrict__ out, int R, int C) {
  __shared__ float tile[32][33];
  int b = blockIdx.z;
  int r0 = blockIdx.y * 32, c0 = blockIdx.x * 32;
  int tx = threadIdx.x & 31, ty = threadIdx.x >> 5;
  const float* ib = in + (long long)b * R * C;
  unsigned short* ob = out + (long long)b * R * C;
  for (int rr = ty; rr < 32; rr += 8)
    tile[rr][tx] = ib[(long long)(r0 + rr) * C + c0 + tx];
  __syncthreads();
  for (int cc = ty; cc < 32; cc += 8)
    ob[(long long)(c0 + cc) * R + r0 + tx] = f32_to_bf16(tile[tx][cc]);
}

__global__ __launch_bounds__(256) void rmsnorm_bf16_kernel(
    const float* __restrict__ x, const float* __restrict__ w,
    unsigned short* __restrict__ out, int ncols) {
  int row = blockIdx.x;
  const float* xr = x + (long long)row * ncols;
  float ss = 0.0f;
  for (int c = threadIdx.x; c < ncols; c += 256) { float v = xr[c]; ss += v * v; }
  for (int m = 16; m >= 1; m >>= 1) ss += __shfl_xor(ss, m, 32);
  __shared__ float red[8];
  if ((threadIdx.x & 31) == 0) red[threadIdx.x >> 5] = ss;
  __syncthreads();
  float tot = 0.0f;
  for (int i = 0; i < 8; ++i) tot += red[i];
  float inv = rsqrtf(tot / (float)ncols + 1e-6f);
  for (int c = threadIdx.x; c < ncols; c += 256)
    out[(long long)row * ncols + c] = f32_to_bf16(xr[c] * inv * w[c]);
}

// ckv_full[T][576] -> kf[T][576]: rmsnorm(0..511) + rope(512..575)
__global__ __launch_bounds__(256) void ckv_kernel(
    const float* __restrict__ ckv, const float* __restrict__ w,
    const long long* __restrict__ pos, unsigned short* __restrict__ kf) {
  int row = blockIdx.x;
  const float* xr = ckv + (long long)row * DF;
  float ss = 0.0f;
  for (int c = threadIdx.x; c < KVR; c += 256) { float v = xr[c]; ss += v * v; }
  for (int m = 16; m >= 1; m >>= 1) ss += __shfl_xor(ss, m, 32);
  __shared__ float red[8];
  if ((threadIdx.x & 31) == 0) red[threadIdx.x >> 5] = ss;
  __syncthreads();
  float tot = 0.0f;
  for (int i = 0; i < 8; ++i) tot += red[i];
  float inv = rsqrtf(tot / (float)KVR + 1e-6f);
  unsigned short* kr = kf + (long long)row * DF;
  for (int c = threadIdx.x; c < KVR; c += 256) kr[c] = f32_to_bf16(xr[c] * inv * w[c]);
  if (threadIdx.x < 32) {
    int i = threadIdx.x;
    float p = (float)pos[row];
    float invf = __expf(-(float)i * 0.28782313662425574f);   // ln(1e4)/32
    float x1 = xr[KVR + i], x2 = xr[KVR + 32 + i];
    float s_, c_; __sincosf(p * invf, &s_, &c_);
    kr[KVR + i]      = f32_to_bf16(x1 * c_ - x2 * s_);
    kr[KVR + 32 + i] = f32_to_bf16(x2 * c_ + x1 * s_);
  }
}

// rope q_pe: q_raw[T][6144] -> qf[h][t][512..575]
__global__ __launch_bounds__(256) void rope_q_kernel(
    const float* __restrict__ q, const long long* __restrict__ pos,
    unsigned short* __restrict__ qf) {
  int t = blockIdx.x;
  float p = (float)pos[t];
  for (int idx = threadIdx.x; idx < NH * 32; idx += 256) {
    int h = idx >> 5, i = idx & 31;
    const float* src = q + (long long)t * QBD + h * QH + DN;
    float x1 = src[i], x2 = src[32 + i];
    float invf = __expf(-(float)i * 0.28782313662425574f);
    float s_, c_; __sincosf(p * invf, &s_, &c_);
    unsigned short* dst = qf + ((long long)h * T_SEQ + t) * DF + KVR;
    dst[i]      = f32_to_bf16(x1 * c_ - x2 * s_);
    dst[32 + i] = f32_to_bf16(x2 * c_ + x1 * s_);
  }
}

// ---------------------------------------------------------------------------
extern "C" void kernel_launch(void* const* d_in, const int* in_sizes, int n_in,
                              void* d_out, int out_size, void* d_ws, size_t ws_size,
                              hipStream_t stream) {
  (void)in_sizes; (void)n_in; (void)out_size; (void)ws_size;
  const float*     h        = (const float*)d_in[0];
  const long long* pos      = (const long long*)d_in[1];
  const float*     w_q_a    = (const float*)d_in[2];
  const float*     q_a_ln_w = (const float*)d_in[3];
  const float*     w_q_b    = (const float*)d_in[4];
  const float*     w_kv_a   = (const float*)d_in[5];
  const float*     kv_a_ln  = (const float*)d_in[6];
  const float*     w_kc     = (const float*)d_in[7];
  const float*     w_vc     = (const float*)d_in[8];
  const float*     w_o      = (const float*)d_in[9];
  float* out = (float*)d_out;

  char* p = (char*)d_ws;
  auto alloc = [&](size_t bytes) {
    char* r = p; p += (bytes + 255) & ~(size_t)255; return r;
  };
  unsigned short* h_bf    = (unsigned short*)alloc((size_t)T_SEQ * HID * 2);
  unsigned short* wqa_bf  = (unsigned short*)alloc((size_t)QR * HID * 2);
  unsigned short* wkva_bf = (unsigned short*)alloc((size_t)DF * HID * 2);
  unsigned short* wqb_bf  = (unsigned short*)alloc((size_t)QBD * QR * 2);
  unsigned short* wo_bf   = (unsigned short*)alloc((size_t)HID * ODIM * 2);
  unsigned short* wkc_t   = (unsigned short*)alloc((size_t)NH * KVR * DN * 2);
  unsigned short* wvc_t   = (unsigned short*)alloc((size_t)NH * DV * KVR * 2);
  float*          qa_raw  = (float*)alloc((size_t)T_SEQ * QR * 4);
  unsigned short* qa_bf   = (unsigned short*)alloc((size_t)T_SEQ * QR * 2);
  float*          ckv_raw = (float*)alloc((size_t)T_SEQ * DF * 4);
  unsigned short* kf      = (unsigned short*)alloc((size_t)T_SEQ * DF * 2);
  float*          q_raw   = (float*)alloc((size_t)T_SEQ * QBD * 4);
  unsigned short* q_bf    = (unsigned short*)alloc((size_t)T_SEQ * QBD * 2);
  unsigned short* qf      = (unsigned short*)alloc((size_t)NH * T_SEQ * DF * 2);
  unsigned short* v_t     = (unsigned short*)alloc((size_t)NH * DV * T_SEQ * 2);
  unsigned short* o_bf    = (unsigned short*)alloc((size_t)T_SEQ * ODIM * 2);

  // ---- bf16 conversions of activations & weights ----
  cvt_bf16_kernel<<<2048, 256, 0, stream>>>(h,      h_bf,    (long long)T_SEQ * HID);
  cvt_bf16_kernel<<<2048, 256, 0, stream>>>(w_q_a,  wqa_bf,  (long long)QR * HID);
  cvt_bf16_kernel<<<2048, 256, 0, stream>>>(w_kv_a, wkva_bf, (long long)DF * HID);
  cvt_bf16_kernel<<<2048, 256, 0, stream>>>(w_q_b,  wqb_bf,  (long long)QBD * QR);
  cvt_bf16_kernel<<<2048, 256, 0, stream>>>(w_o,    wo_bf,   (long long)HID * ODIM);
  transpose_cvt_kernel<<<dim3(KVR / 32, DN / 32, NH), 256, 0, stream>>>(w_kc, wkc_t, DN, KVR);
  transpose_cvt_kernel<<<dim3(DV / 32, KVR / 32, NH), 256, 0, stream>>>(w_vc, wvc_t, KVR, DV);

  // ---- q_a = h @ w_q_a^T  (2048x1536x5120), ckv = h @ w_kv_a^T ----
  gemm_nt_kernel<0><<<dim3((QR + 127) / 128, T_SEQ / 128, 1), 256, 0, stream>>>(
      h_bf, HID, 0, wqa_bf, HID, 0, qa_raw, QR, 0, QR, HID);
  gemm_nt_kernel<0><<<dim3((DF + 127) / 128, T_SEQ / 128, 1), 256, 0, stream>>>(
      h_bf, HID, 0, wkva_bf, HID, 0, ckv_raw, DF, 0, DF, HID);

  // ---- norms + k rope ----
  rmsnorm_bf16_kernel<<<T_SEQ, 256, 0, stream>>>(qa_raw, q_a_ln_w, qa_bf, QR);
  ckv_kernel<<<T_SEQ, 256, 0, stream>>>(ckv_raw, kv_a_ln, pos, kf);

  // ---- q = rms(q_a) @ w_q_b^T  (2048x6144x1536) ----
  gemm_nt_kernel<0><<<dim3(QBD / 128, T_SEQ / 128, 1), 256, 0, stream>>>(
      qa_bf, QR, 0, wqb_bf, QR, 0, q_raw, QBD, 0, QBD, QR);
  cvt_bf16_kernel<<<2048, 256, 0, stream>>>(q_raw, q_bf, (long long)T_SEQ * QBD);
  rope_q_kernel<<<T_SEQ, 256, 0, stream>>>(q_raw, pos, qf);

  // ---- q_lat[h] = q_nope[h] @ w_kc[h]  -> qf[h][:, 0:512] bf16 (batched) ----
  gemm_nt_kernel<1><<<dim3(KVR / 128, T_SEQ / 128, NH), 256, 0, stream>>>(
      q_bf, QBD, QH,                          // A: per-head column offset h*192
      wkc_t, DN, (long long)KVR * DN,
      qf, DF, (long long)T_SEQ * DF, KVR, DN);

  // ---- V[h] = ckv @ w_vc[h] -> v_t[h][v][t] bf16 transposed (batched) ----
  gemm_nt_kernel<2><<<dim3(DV / 128, T_SEQ / 128, NH), 256, 0, stream>>>(
      kf, DF, 0,                              // rmsnormed ckv, lda=576 (cols 0..511)
      wvc_t, KVR, (long long)DV * KVR,
      v_t, T_SEQ, (long long)DV * T_SEQ, DV, KVR);

  // ---- flash attention ----
  attn_kernel<<<dim3(T_SEQ / 32, NH), 64, 0, stream>>>(qf, kf, v_t, o_bf);

  // ---- out = O @ w_o^T  (2048x5120x4096) ----
  gemm_nt_kernel<0><<<dim3(HID / 128, T_SEQ / 128, 1), 256, 0, stream>>>(
      o_bf, ODIM, 0, wo_bf, ODIM, 0, out, HID, 0, HID, ODIM);
}
